// ULREmbedding_31903017075210
// MI455X (gfx1250) — matrix operand order repaired and checked by reading, not compiled, not run on hardware
//
#include <hip/hip_runtime.h>
#include <math.h>

// ---------------- problem constants ----------------
#define EMB     512
#define NROWS   2048          // B*M = 8*256
#define UVOCAB  50000
#define INVT    20.0f         // 1 / TEMPERATURE (folded into bf16 Q)
// ---------------- tiling ----------------
#define ROWS_WG 32
#define USTEP   64
#define BLOCK   256           // 8 waves of 32
#define SSTRIDE 68            // f32 stride for S tile (16B-aligned rows, bank-skewed)
#define PSTRIDE 72            // bf16 stride for P tile (16B-aligned rows)
#define VTSTRIDE 72           // bf16 stride for V^T tile
#define PREC    516           // floats per (row,chunk) partial: 512 O + m + l + pad
#define MAXNC   16

typedef __attribute__((ext_vector_type(16))) __bf16 v16bf;
typedef __attribute__((ext_vector_type(8)))  float  v8f;

union BFrag { v16bf v; unsigned short u[16]; uint4 q[2]; };
union CFrag { v8f v; float f[8]; };

// native f32 -> bf16 (RNE); lowers to hardware cvt on gfx1250
__device__ __forceinline__ unsigned short f2bf(float x) {
  __bf16 h = (__bf16)x;
  return __builtin_bit_cast(unsigned short, h);
}
__device__ __forceinline__ unsigned pk2bf(float lo, float hi) {
  return (unsigned)f2bf(lo) | ((unsigned)f2bf(hi) << 16);
}

// ---------------------------------------------------------------------------
// Kernel A: bulk f32 -> bf16 conversion (for K and V tables). n % 8 == 0.
// ---------------------------------------------------------------------------
__global__ void cvt_kernel(const float* __restrict__ s,
                           unsigned short* __restrict__ d, long n) {
  long i = ((long)blockIdx.x * BLOCK + threadIdx.x) * 8;
  if (i >= n) return;
  float4 a = *(const float4*)(s + i);
  float4 b = *(const float4*)(s + i + 4);
  uint4 o;
  o.x = pk2bf(a.x, a.y);
  o.y = pk2bf(a.z, a.w);
  o.z = pk2bf(b.x, b.y);
  o.w = pk2bf(b.z, b.w);
  *(uint4*)(d + i) = o;
}

// ---------------------------------------------------------------------------
// Kernel 0: q = gather(src_query_emb)[2048,512] @ transform * INVT -> bf16 ws
// ---------------------------------------------------------------------------
__global__ void qgen_kernel(const int* __restrict__ tok,
                            const float* __restrict__ sq,
                            const float* __restrict__ T,
                            unsigned short* __restrict__ qbf) {
  __shared__ float qs[16 * 512];
  __shared__ int   ts[16];
  const int rb = blockIdx.x * 16;
  const int tid = threadIdx.x;
  if (tid < 16) ts[tid] = tok[rb + tid];
  __syncthreads();
  for (int i = tid; i < 16 * 512; i += BLOCK) {
    int r = i >> 9, e = i & 511;
    qs[i] = sq[(long)ts[r] * EMB + e];
  }
  __syncthreads();
  float acc0[16], acc1[16];
#pragma unroll
  for (int r = 0; r < 16; ++r) { acc0[r] = 0.f; acc1[r] = 0.f; }
  for (int k = 0; k < EMB; ++k) {
    float t0 = T[(long)k * EMB + tid];
    float t1 = T[(long)k * EMB + tid + 256];
#pragma unroll
    for (int r = 0; r < 16; ++r) {
      float qv = qs[r * 512 + k];
      acc0[r] = fmaf(qv, t0, acc0[r]);
      acc1[r] = fmaf(qv, t1, acc1[r]);
    }
  }
#pragma unroll
  for (int r = 0; r < 16; ++r) {
    qbf[(long)(rb + r) * EMB + tid]       = f2bf(acc0[r] * INVT);
    qbf[(long)(rb + r) * EMB + tid + 256] = f2bf(acc1[r] * INVT);
  }
}

// ---------------------------------------------------------------------------
// Kernel 1: flash-attention over a vocab chunk.
//   grid = (NROWS/ROWS_WG, nchunk). 8 waves of 32.
//   S-phase: wave w -> (mi = w>>2 row-frag, ni = w&3 col-frag of 64-wide u-step)
//   O-phase: wave w owns emb columns [w*64, w*64+64) : 2x4 C fragments.
// BF16KV=true: keys/vals already bf16 in workspace (no in-loop conversion).
// ---------------------------------------------------------------------------
template <bool BF16KV>
__global__ void attn_kernel(const unsigned short* __restrict__ qbf,
                            const void* __restrict__ keysp,
                            const void* __restrict__ valsp,
                            float* __restrict__ part,
                            int uchunk) {
  extern __shared__ char smem[];
  float*          S        = (float*)smem;                                 // 32 x SSTRIDE f32
  unsigned short* P        = (unsigned short*)(smem + 32 * SSTRIDE * 4);   // 32 x PSTRIDE bf16
  unsigned short* VT       = P + 32 * PSTRIDE;                             // 512 x VTSTRIDE bf16
  float*          rowscale = (float*)(VT + 512 * VTSTRIDE);                // 32 f32

  const int tid     = threadIdx.x;
  const int wave    = tid >> 5;
  const int lane    = tid & 31;
  const int half    = lane >> 4;
  const int lanelow = lane & 15;

  const int  row0    = blockIdx.x * ROWS_WG;
  const long u_begin = (long)blockIdx.y * uchunk;
  long u_end = u_begin + uchunk; if (u_end > UVOCAB) u_end = UVOCAB;
  const long ulast = u_end - 1;

  const int mi = wave >> 2;
  const int ni = wave & 3;

  CFrag Oacc[2][4];
#pragma unroll
  for (int m2 = 0; m2 < 2; ++m2)
#pragma unroll
    for (int n = 0; n < 4; ++n)
#pragma unroll
      for (int r = 0; r < 8; ++r) Oacc[m2][n].f[r] = 0.f;

  float m_run = -__builtin_inff();
  float l_run = 0.f;
  const int srow = tid >> 3;   // softmax row 0..31 (8 threads per row)
  const int sseg = tid & 7;

  const unsigned short* abase0 =
      qbf + (long)(row0 + mi * 16 + lanelow) * EMB + half * 16;

  for (long u0 = u_begin; u0 < u_end; u0 += USTEP) {
    __syncthreads();  // protect previous iteration's P / VT consumers

    // ---- stage V^T tile (USTEP x EMB, transposed, bf16) into LDS ----
    if constexpr (BF16KV) {
      const unsigned short* vb = (const unsigned short*)valsp;
#pragma unroll 2
      for (int i = tid; i < (USTEP / 2) * EMB; i += BLOCK) {
        int e  = i & 511;
        int up = (i >> 9) * 2;
        long ua = u0 + up;
        long ua0 = ua     <= ulast ? ua     : ulast;
        long ua1 = ua + 1 <= ulast ? ua + 1 : ulast;
        unsigned v0 = vb[ua0 * EMB + e];
        unsigned v1 = vb[ua1 * EMB + e];
        *(unsigned int*)&VT[(long)e * VTSTRIDE + up] = v0 | (v1 << 16);
      }
    } else {
      const float* vf = (const float*)valsp;
#pragma unroll 2
      for (int i = tid; i < (USTEP / 2) * EMB; i += BLOCK) {
        int e  = i & 511;
        int up = (i >> 9) * 2;
        long ua = u0 + up;
        long ua0 = ua     <= ulast ? ua     : ulast;
        long ua1 = ua + 1 <= ulast ? ua + 1 : ulast;
        *(unsigned int*)&VT[(long)e * VTSTRIDE + up] =
            pk2bf(vf[ua0 * EMB + e], vf[ua1 * EMB + e]);
      }
    }

    // ---- S tile: S[32,64] = Qtile x K^T (each wave one 16x16 tile) ----
    const long ucol   = u0 + ni * 16 + lanelow;      // this lane's key row (B column)
    const long uclamp = ucol <= ulast ? ucol : ulast;

    CFrag sc;
#pragma unroll
    for (int r = 0; r < 8; ++r) sc.f[r] = 0.f;

    if constexpr (BF16KV) {
      const unsigned short* kb =
          (const unsigned short*)keysp + uclamp * EMB + half * 16;
      __builtin_prefetch(kb + USTEP * EMB, 0, 0);
#pragma unroll 4
      for (int k0 = 0; k0 < EMB; k0 += 32) {
        BFrag A, B;
        A.q[0] = *(const uint4*)(abase0 + k0);
        A.q[1] = *(const uint4*)(abase0 + k0 + 8);
        B.q[0] = *(const uint4*)(kb + k0);
        B.q[1] = *(const uint4*)(kb + k0 + 8);
        sc.v = __builtin_amdgcn_wmma_f32_16x16x32_bf16(
            false, A.v, false, B.v, (short)0, sc.v, false, false);
      }
    } else {
      const float* kbase = (const float*)keysp + uclamp * EMB + half * 16;
#pragma unroll 2
      for (int k0 = 0; k0 < EMB; k0 += 32) {
        BFrag A, B;
        A.q[0] = *(const uint4*)(abase0 + k0);
        A.q[1] = *(const uint4*)(abase0 + k0 + 8);
        const float4* kf = (const float4*)(kbase + k0);
        float4 f0 = kf[0], f1 = kf[1], f2 = kf[2], f3 = kf[3];
        B.q[0].x = pk2bf(f0.x, f0.y); B.q[0].y = pk2bf(f0.z, f0.w);
        B.q[0].z = pk2bf(f1.x, f1.y); B.q[0].w = pk2bf(f1.z, f1.w);
        B.q[1].x = pk2bf(f2.x, f2.y); B.q[1].y = pk2bf(f2.z, f2.w);
        B.q[1].z = pk2bf(f3.x, f3.y); B.q[1].w = pk2bf(f3.z, f3.w);
        sc.v = __builtin_amdgcn_wmma_f32_16x16x32_bf16(
            false, A.v, false, B.v, (short)0, sc.v, false, false);
      }
    }

    const bool colvalid = (ucol < u_end);
#pragma unroll
    for (int r = 0; r < 8; ++r) {
      float s = colvalid ? sc.f[r] : -1e30f;   // INVT already folded into Q
      S[(mi * 16 + half * 8 + r) * SSTRIDE + ni * 16 + lanelow] = s;
    }
    __syncthreads();

    // ---- online softmax: 8 threads per row, 8 columns each ----
    float x[8];
    {
      float4 a = *(const float4*)&S[srow * SSTRIDE + sseg * 8];
      float4 b = *(const float4*)&S[srow * SSTRIDE + sseg * 8 + 4];
      x[0]=a.x; x[1]=a.y; x[2]=a.z; x[3]=a.w;
      x[4]=b.x; x[5]=b.y; x[6]=b.z; x[7]=b.w;
    }
    float mt = x[0];
#pragma unroll
    for (int i = 1; i < 8; ++i) mt = fmaxf(mt, x[i]);
    mt = fmaxf(mt, __shfl_xor(mt, 1, 8));
    mt = fmaxf(mt, __shfl_xor(mt, 2, 8));
    mt = fmaxf(mt, __shfl_xor(mt, 4, 8));
    const float m_new = fmaxf(m_run, mt);
    const float scl   = __expf(m_run - m_new);
    float lsum = 0.f;
    float p[8];
#pragma unroll
    for (int i = 0; i < 8; ++i) { p[i] = __expf(x[i] - m_new); lsum += p[i]; }
    lsum += __shfl_xor(lsum, 1, 8);
    lsum += __shfl_xor(lsum, 2, 8);
    lsum += __shfl_xor(lsum, 4, 8);
    l_run = l_run * scl + lsum;
    m_run = m_new;
#pragma unroll
    for (int i = 0; i < 4; ++i) {
      *(unsigned int*)&P[srow * PSTRIDE + sseg * 8 + 2 * i] =
          pk2bf(p[2 * i], p[2 * i + 1]);
    }
    if (sseg == 0) rowscale[srow] = scl;
    __syncthreads();

    // ---- rescale O accumulators ----
    float scA[8], scB[8];
#pragma unroll
    for (int r = 0; r < 8; ++r) {
      scA[r] = rowscale[half * 8 + r];
      scB[r] = rowscale[16 + half * 8 + r];
    }
#pragma unroll
    for (int n = 0; n < 4; ++n)
#pragma unroll
      for (int r = 0; r < 8; ++r) {
        Oacc[0][n].f[r] *= scA[r];
        Oacc[1][n].f[r] *= scB[r];
      }

    // ---- O += P x V  (A = P from LDS, B = V^T from LDS) ----
    // All 6 fragment loads of a kk-step are issued before the 8 WMMAs so the
    // scheduler can overlap ds_load latency with matrix ops (progressive
    // s_wait_dscnt instead of full serialization).
#pragma unroll
    for (int kk = 0; kk < USTEP; kk += 32) {
      BFrag Ap[2], Bv[4];
#pragma unroll
      for (int m2 = 0; m2 < 2; ++m2) {
        const unsigned short* pb = P + (m2 * 16 + lanelow) * PSTRIDE + kk + half * 16;
        Ap[m2].q[0] = *(const uint4*)pb;
        Ap[m2].q[1] = *(const uint4*)(pb + 8);
      }
#pragma unroll
      for (int n = 0; n < 4; ++n) {
        const unsigned short* vbp =
            VT + (long)(wave * 64 + n * 16 + lanelow) * VTSTRIDE + kk + half * 16;
        Bv[n].q[0] = *(const uint4*)vbp;
        Bv[n].q[1] = *(const uint4*)(vbp + 8);
      }
#pragma unroll
      for (int n = 0; n < 4; ++n)
#pragma unroll
        for (int m2 = 0; m2 < 2; ++m2)
          Oacc[m2][n].v = __builtin_amdgcn_wmma_f32_16x16x32_bf16(
              false, Ap[m2].v, false, Bv[n].v, (short)0, Oacc[m2][n].v, false, false);
    }
  }

  // ---- write per-chunk partials: unnormalized O plus (m, l) per row ----
  float* pbase = part + (long)blockIdx.y * NROWS * PREC;
#pragma unroll
  for (int m2 = 0; m2 < 2; ++m2)
#pragma unroll
    for (int n = 0; n < 4; ++n)
#pragma unroll
      for (int r = 0; r < 8; ++r) {
        int rg = row0 + m2 * 16 + half * 8 + r;
        pbase[(long)rg * PREC + wave * 64 + n * 16 + lanelow] = Oacc[m2][n].f[r];
      }
  if (sseg == 0) {
    int rg = row0 + srow;
    pbase[(long)rg * PREC + 512] = m_run;
    pbase[(long)rg * PREC + 513] = l_run;
  }
}

// ---------------------------------------------------------------------------
// Kernel 2: combine chunk partials + residual.
// ---------------------------------------------------------------------------
__global__ void combine_kernel(const float* __restrict__ part,
                               const int* __restrict__ tok,
                               const float* __restrict__ mono,
                               const float* __restrict__ alpha,
                               float* __restrict__ out,
                               int nchunk) {
  long idx = (long)blockIdx.x * blockDim.x + threadIdx.x;
  int r = (int)(idx >> 9);
  int e = (int)(idx & 511);
  float mmax = -__builtin_inff();
  for (int c = 0; c < nchunk; ++c)
    mmax = fmaxf(mmax, part[((long)c * NROWS + r) * PREC + 512]);
  float L = 0.f, acc = 0.f;
  for (int c = 0; c < nchunk; ++c) {
    long  base = ((long)c * NROWS + r) * PREC;
    float w = __expf(part[base + 512] - mmax);
    L   += w * part[base + 513];
    acc += w * part[base + e];
  }
  float ue = (L > 0.f) ? acc / L : 0.f;
  int t = tok[r];
  out[idx] = alpha[t] * mono[(long)t * EMB + e] + ue;
}

// ---------------------------------------------------------------------------
extern "C" void kernel_launch(void* const* d_in, const int* in_sizes, int n_in,
                              void* d_out, int out_size, void* d_ws, size_t ws_size,
                              hipStream_t stream) {
  (void)in_sizes; (void)n_in; (void)out_size;
  const int*   tok   = (const int*)d_in[0];
  const float* T     = (const float*)d_in[1];
  const float* sq    = (const float*)d_in[2];
  const float* mono  = (const float*)d_in[3];
  const float* keys  = (const float*)d_in[4];
  const float* vals  = (const float*)d_in[5];
  const float* alpha = (const float*)d_in[6];
  float* out = (float*)d_out;

  char* base = (char*)d_ws;
  unsigned short* qbf = (unsigned short*)base;
  const size_t qbytes    = (size_t)NROWS * EMB * 2;
  const size_t kvbytes   = (size_t)UVOCAB * EMB * 2;     // each table, bf16
  const size_t per_chunk = (size_t)NROWS * PREC * 4;

  const bool bf16kv = (ws_size >= qbytes + 2 * kvbytes + per_chunk);

  unsigned short* kbf = (unsigned short*)(base + qbytes);
  unsigned short* vbf = kbf + (size_t)UVOCAB * EMB;

  float* part;
  int nchunk;
  if (bf16kv) {
    part = (float*)(base + qbytes + 2 * kvbytes);
    size_t n = (ws_size - qbytes - 2 * kvbytes) / per_chunk;
    nchunk = (n > MAXNC) ? MAXNC : (int)n;
  } else {
    part = (float*)(base + qbytes);
    size_t avail = (ws_size > qbytes) ? ws_size - qbytes : 0;
    size_t n = avail / per_chunk;
    nchunk = (n < 1) ? 1 : ((n > MAXNC) ? MAXNC : (int)n);
  }
  const int uchunk = (UVOCAB + nchunk - 1) / nchunk;

  qgen_kernel<<<NROWS / 16, BLOCK, 0, stream>>>(tok, sq, T, qbf);

  const size_t lds = 32 * SSTRIDE * 4 + 32 * PSTRIDE * 2 + 512 * VTSTRIDE * 2 + 32 * 4;
  dim3 grid(NROWS / ROWS_WG, nchunk);

  if (bf16kv) {
    const long kn = (long)UVOCAB * EMB;                  // 25,600,000 (mult of 8)
    const int cblk = (int)((kn / 8 + BLOCK - 1) / BLOCK);
    cvt_kernel<<<cblk, BLOCK, 0, stream>>>(keys, kbf, kn);
    cvt_kernel<<<cblk, BLOCK, 0, stream>>>(vals, vbf, kn);
    attn_kernel<true><<<grid, BLOCK, lds, stream>>>(qbf, kbf, vbf, part, uchunk);
  } else {
    attn_kernel<false><<<grid, BLOCK, lds, stream>>>(qbf, keys, vals, part, uchunk);
  }

  combine_kernel<<<(NROWS * EMB) / BLOCK, BLOCK, 0, stream>>>(part, tok, mono, alpha, out, nchunk);
}
